// GraphAttentionLayer_84610855731510
// MI455X (gfx1250) — matrix operand order repaired and checked by reading, not compiled, not run on hardware
//
#include <hip/hip_runtime.h>
#include <hip/hip_bf16.h>

// ---- sizes (match reference) ----
#define NPTS 8192
#define CCH  128
#define KNN  16
#define CS   16      // C / S
#define BN_RS 0.9999950000374997f   // 1/sqrt(1 + 1e-5)

typedef __attribute__((ext_vector_type(16))) _Float16 v16h;
typedef __attribute__((ext_vector_type(8)))  _Float16 v8h;
typedef __attribute__((ext_vector_type(8)))  float    v8f;

union V16U { v16h v; v8h h[2]; };

// ============================================================
// 1) f32 -> f16 convert (activations)
// ============================================================
__global__ void cvt_f16_kernel(const float* __restrict__ in, _Float16* __restrict__ out, int n) {
    int i = blockIdx.x * blockDim.x + threadIdx.x;
    if (i < n) out[i] = (_Float16)in[i];
}

// in: [rows][cols] f32 row-major  ->  out: [cols][rows] f16 (i.e. W^T)
__global__ void transpose_f16_kernel(const float* __restrict__ in, _Float16* __restrict__ out,
                                     int rows, int cols) {
    int i = blockIdx.x * blockDim.x + threadIdx.x;
    if (i < rows * cols) {
        int r = i / cols, c = i - r * cols;
        out[c * rows + r] = (_Float16)in[i];
    }
}

// ============================================================
// 2) KNN: per-thread query, candidates staged via async global->LDS copy
//    (GLOBAL_LOAD_ASYNC_TO_LDS_B32, tracked by ASYNCcnt), top-16 insertion.
// ============================================================
__global__ void __launch_bounds__(256)
knn_kernel(const float* __restrict__ p, int* __restrict__ idx, float* __restrict__ dist) {
    const int q = blockIdx.x * 256 + threadIdx.x;
    const float qx = p[q * 3 + 0], qy = p[q * 3 + 1], qz = p[q * 3 + 2];

    float bd[KNN];
    int   bi[KNN];
#pragma unroll
    for (int i = 0; i < KNN; ++i) { bd[i] = 3.0e38f; bi[i] = 0; }

    __shared__ float sp[256 * 3];
    for (int base = 0; base < NPTS; base += 256) {
        const int j = base + threadIdx.x;
        {
            // async copy 3 dwords of this thread's candidate point straight to LDS
            const float* g  = &p[(size_t)j * 3];
            // low 32 bits of a generic LDS address = group-segment byte offset (ISA 10.2)
            uint32_t l0 = (uint32_t)(uintptr_t)&sp[threadIdx.x * 3];
            asm volatile("global_load_async_to_lds_b32 %0, %1, off" :: "v"(l0),      "v"(g)     : "memory");
            asm volatile("global_load_async_to_lds_b32 %0, %1, off" :: "v"(l0 + 4u), "v"(g + 1) : "memory");
            asm volatile("global_load_async_to_lds_b32 %0, %1, off" :: "v"(l0 + 8u), "v"(g + 2) : "memory");
            asm volatile("s_wait_asynccnt 0" ::: "memory");
        }
        __syncthreads();
        for (int t = 0; t < 256; ++t) {
            float dx = sp[t * 3 + 0] - qx;
            float dy = sp[t * 3 + 1] - qy;
            float dz = sp[t * 3 + 2] - qz;
            float d  = dx * dx + dy * dy + dz * dz;
            if (d < bd[KNN - 1]) {                 // stable ascending insert (ties keep lower idx)
                int pos = KNN - 1;
                while (pos > 0 && bd[pos - 1] > d) {
                    bd[pos] = bd[pos - 1]; bi[pos] = bi[pos - 1]; --pos;
                }
                bd[pos] = d; bi[pos] = base + t;
            }
        }
        __syncthreads();
    }
#pragma unroll
    for (int i = 0; i < KNN; ++i) {
        idx[q * KNN + i]  = bi[i];
        dist[q * KNN + i] = sqrtf(fmaxf(bd[i], 0.0f));
    }
}

// ============================================================
// 3) Fused Q/K/V projection GEMM: one wave owns a 16x128 output strip of one
//    of the three matrices -> A fragment loaded once per k-step, reused for
//    all 8 B tiles (32 WMMAs per wave). Bt* = W^T as f16 so B fragments are
//    single contiguous 32B loads.
// ============================================================
__global__ void __launch_bounds__(256)
gemm_proj_kernel(const _Float16* __restrict__ Ah,
                 const _Float16* __restrict__ Btq, const _Float16* __restrict__ Btk,
                 const _Float16* __restrict__ Btv,
                 const float* __restrict__ bq, const float* __restrict__ bk,
                 const float* __restrict__ bv,
                 float* __restrict__ Oq, float* __restrict__ Ok, float* __restrict__ Ov) {
    const int wave = (blockIdx.x * 256 + threadIdx.x) >> 5;   // 1536 waves
    const int lane = threadIdx.x & 31;
    const int mat   = wave / (NPTS / 16);                     // 0..2 (q/k/v), wave-uniform
    const int strip = wave - mat * (NPTS / 16);               // 0..511
    const int m0 = strip * 16;
    const int half = lane >> 4, r = lane & 15;

    const _Float16* Bt   = (mat == 0) ? Btq : (mat == 1 ? Btk : Btv);
    const float*    bias = (mat == 0) ? bq  : (mat == 1 ? bk  : bv);
    float*          Out  = (mat == 0) ? Oq  : (mat == 1 ? Ok  : Ov);

    v8f acc[8] = {};
    const _Float16* arow = Ah + (size_t)(m0 + r) * CCH + half * 16;  // row M=r
#pragma unroll
    for (int k0 = 0; k0 < CCH; k0 += 32) {
        V16U ua;
        ua.h[0] = *(const v8h*)(arow + k0);        // e=0..7  -> K = k0+half*16+e
        ua.h[1] = *(const v8h*)(arow + k0 + 8);    // e=8..15 -> K = k0+half*16+e
#pragma unroll
        for (int tn = 0; tn < 8; ++tn) {
            v16h b = *(const v16h*)(Bt + (size_t)(tn * 16 + r) * CCH + k0 + half * 16);
            acc[tn] = __builtin_amdgcn_wmma_f32_16x16x32_f16(false, ua.v, false, b,
                                                             (short)0, acc[tn], false, false);
        }
    }
#pragma unroll
    for (int tn = 0; tn < 8; ++tn) {
        const float b = bias[tn * 16 + r];
        float* ocol = Out + (size_t)m0 * CCH + tn * 16 + r;
#pragma unroll
        for (int v = 0; v < 8; ++v)                // D: acc[v] = (M=half*8+v, N=r)
            ocol[(size_t)(half * 8 + v) * CCH] = acc[tn][v] + b;
    }
}

// ============================================================
// 4) Fused attention: one wave per point (4 waves / 128-thread block).
//    pr -> LDS, WMMA for w@Ww1 (16x16x128), scalar 16x16 for Ww2,
//    cross-half softmax over neighbors, distance-weighted grouped sum.
// ============================================================
__global__ void __launch_bounds__(128)
attn_kernel(const float* __restrict__ p,
            const float* __restrict__ Qp, const float* __restrict__ Kp, const float* __restrict__ Vp,
            const int* __restrict__ idx, const float* __restrict__ dist,
            const float* __restrict__ Wp1, const float* __restrict__ bp1,
            const float* __restrict__ g1,  const float* __restrict__ be1,
            const float* __restrict__ Wp2, const float* __restrict__ bp2,
            const float* __restrict__ g2,  const float* __restrict__ be2,
            const _Float16* __restrict__ Ww1t, const float* __restrict__ bw1,
            const float* __restrict__ g3,  const float* __restrict__ be3,
            const float* __restrict__ Ww2, const float* __restrict__ bw2,
            float* __restrict__ Out) {
    const int w    = threadIdx.x >> 5;            // wave in block: 0..3
    const int lane = threadIdx.x & 31;
    const int half = lane >> 4, r = lane & 15;
    const int n    = blockIdx.x * 4 + w;

    __shared__ float s_pr[4][KNN * CCH];          // 32 KB: pr[k][c]
    __shared__ float s_h3[4][KNN * 4];            // hidden 3-vec of linear_p
    __shared__ float s_w1[4][KNN * CS];           // relu(bn3(w@Ww1+bw1))
    __shared__ float s_wt[4][KNN * CS];           // softmax weights
    __shared__ int   s_idx[4][KNN];
    __shared__ float s_expd[4][KNN];

    if (lane < KNN) {
        s_idx[w][lane]  = idx[n * KNN + lane];
        s_expd[w][lane] = __expf(-dist[n * KNN + lane]);
    }
    __syncthreads();

    // ---- linear_p stage 1: h = relu(bn1(p_r @ Wp1 + bp1)), per neighbor ----
    if (lane < KNN) {
        const int nb = s_idx[w][lane];
        float rx = p[nb * 3 + 0] - p[n * 3 + 0];
        float ry = p[nb * 3 + 1] - p[n * 3 + 1];
        float rz = p[nb * 3 + 2] - p[n * 3 + 2];
#pragma unroll
        for (int j = 0; j < 3; ++j) {
            float t = rx * Wp1[0 * 3 + j] + ry * Wp1[1 * 3 + j] + rz * Wp1[2 * 3 + j] + bp1[j];
            t = t * (g1[j] * BN_RS) + be1[j];
            s_h3[w][lane * 4 + j] = fmaxf(t, 0.0f);
        }
    }
    __syncthreads();

    // ---- linear_p stage 2: pr[k][c] = h[k] @ Wp2 + bp2 ----
    for (int i = lane; i < KNN * CCH; i += 32) {
        int k = i >> 7, c = i & (CCH - 1);
        float v = s_h3[w][k * 4 + 0] * Wp2[0 * CCH + c]
                + s_h3[w][k * 4 + 1] * Wp2[1 * CCH + c]
                + s_h3[w][k * 4 + 2] * Wp2[2 * CCH + c] + bp2[c];
        s_pr[w][i] = v;
    }
    __syncthreads();

    // ---- GEMM1: [16 nbrs x 128] @ Ww1 -> [16 x 16], WMMA, A built on the fly ----
    const int nbr = s_idx[w][r];
    const float* qrow = Qp + (size_t)n   * CCH;
    const float* krow = Kp + (size_t)nbr * CCH;
    v8f acc = {};
#pragma unroll
    for (int k0 = 0; k0 < CCH; k0 += 32) {
        v16h a;
#pragma unroll
        for (int e = 0; e < 16; ++e) {
            int c = k0 + half * 16 + e;
            float t = qrow[c] - krow[c] + s_pr[w][r * CCH + c];
            t = t * (g2[c] * BN_RS) + be2[c];       // bn2
            a[e] = (_Float16)fmaxf(t, 0.0f);        // relu
        }
        v16h b = *(const v16h*)(Ww1t + (size_t)r * CCH + k0 + half * 16);
        acc = __builtin_amdgcn_wmma_f32_16x16x32_f16(false, a, false, b,
                                                     (short)0, acc, false, false);
    }
    {   // D: acc[v] = (m = half*8+v, j = r); apply bw1, bn3, relu -> LDS
        const float sc = g3[r] * BN_RS, sh = be3[r], bb = bw1[r];
#pragma unroll
        for (int v = 0; v < 8; ++v) {
            float t = (acc[v] + bb) * sc + sh;
            s_w1[w][(half * 8 + v) * CS + r] = fmaxf(t, 0.0f);
        }
    }
    __syncthreads();

    // ---- GEMM2 (scalar 16x16x16) + softmax over neighbors (axis m) ----
    float sco[8];
#pragma unroll
    for (int v = 0; v < 8; ++v) {
        const int m = half * 8 + v;
        float s = bw2[r];
#pragma unroll
        for (int j = 0; j < CS; ++j)
            s += s_w1[w][m * CS + j] * Ww2[j * CS + r];
        sco[v] = s;
    }
    float mx = sco[0];
#pragma unroll
    for (int v = 1; v < 8; ++v) mx = fmaxf(mx, sco[v]);
    mx = fmaxf(mx, __shfl_xor(mx, 16));             // combine halves (m=0..7 | 8..15)
    float sum = 0.0f;
#pragma unroll
    for (int v = 0; v < 8; ++v) { sco[v] = __expf(sco[v] - mx); sum += sco[v]; }
    sum += __shfl_xor(sum, 16);
    const float inv = 1.0f / sum;
#pragma unroll
    for (int v = 0; v < 8; ++v)
        s_wt[w][(half * 8 + v) * CS + r] = sco[v] * inv;
    __syncthreads();

    // ---- out[n][c] = sum_k (V[idx[k]][c]*exp(-d[k]) + pr[k][c]) * wt[k][c&15] ----
#pragma unroll
    for (int t = 0; t < 4; ++t) {
        const int c = t * 32 + lane;
        const int j = c & (CS - 1);
        float o = 0.0f;
#pragma unroll
        for (int k = 0; k < KNN; ++k) {
            const int nb = s_idx[w][k];
            float vv = Vp[(size_t)nb * CCH + c] * s_expd[w][k] + s_pr[w][k * CCH + c];
            o += vv * s_wt[w][k * CS + j];
        }
        Out[(size_t)n * CCH + c] = o;
    }
}

// ============================================================
// launch
// ============================================================
extern "C" void kernel_launch(void* const* d_in, const int* in_sizes, int n_in,
                              void* d_out, int out_size, void* d_ws, size_t ws_size,
                              hipStream_t stream) {
    const float* p   = (const float*)d_in[0];
    const float* x   = (const float*)d_in[1];
    const float* Wq  = (const float*)d_in[2];  const float* bq  = (const float*)d_in[3];
    const float* Wk  = (const float*)d_in[4];  const float* bk  = (const float*)d_in[5];
    const float* Wv  = (const float*)d_in[6];  const float* bv  = (const float*)d_in[7];
    const float* Wp1 = (const float*)d_in[8];  const float* bp1 = (const float*)d_in[9];
    const float* g1  = (const float*)d_in[10]; const float* be1 = (const float*)d_in[11];
    const float* Wp2 = (const float*)d_in[12]; const float* bp2 = (const float*)d_in[13];
    const float* g2  = (const float*)d_in[14]; const float* be2 = (const float*)d_in[15];
    const float* Ww1 = (const float*)d_in[16]; const float* bw1 = (const float*)d_in[17];
    const float* g3  = (const float*)d_in[18]; const float* be3 = (const float*)d_in[19];
    const float* Ww2 = (const float*)d_in[20]; const float* bw2 = (const float*)d_in[21];
    float* out = (float*)d_out;

    // ---- workspace layout ----
    char* ws = (char*)d_ws;
    size_t off = 0;
    auto alloc = [&](size_t bytes) { char* r = ws + off; off = (off + bytes + 255) & ~(size_t)255; return r; };
    _Float16* xh    = (_Float16*)alloc((size_t)NPTS * CCH * 2);
    _Float16* Wqt_h = (_Float16*)alloc((size_t)CCH * CCH * 2);
    _Float16* Wkt_h = (_Float16*)alloc((size_t)CCH * CCH * 2);
    _Float16* Wvt_h = (_Float16*)alloc((size_t)CCH * CCH * 2);
    _Float16* Ww1t  = (_Float16*)alloc((size_t)CS * CCH * 2);
    float*    Qp    = (float*)alloc((size_t)NPTS * CCH * 4);
    float*    Kp    = (float*)alloc((size_t)NPTS * CCH * 4);
    float*    Vp    = (float*)alloc((size_t)NPTS * CCH * 4);
    int*      idx   = (int*)alloc((size_t)NPTS * KNN * 4);
    float*    dist  = (float*)alloc((size_t)NPTS * KNN * 4);

    // 1) precision conversion + weight transposes
    cvt_f16_kernel<<<(NPTS * CCH + 255) / 256, 256, 0, stream>>>(x, xh, NPTS * CCH);
    transpose_f16_kernel<<<(CCH * CCH + 255) / 256, 256, 0, stream>>>(Wq, Wqt_h, CCH, CCH);
    transpose_f16_kernel<<<(CCH * CCH + 255) / 256, 256, 0, stream>>>(Wk, Wkt_h, CCH, CCH);
    transpose_f16_kernel<<<(CCH * CCH + 255) / 256, 256, 0, stream>>>(Wv, Wvt_h, CCH, CCH);
    transpose_f16_kernel<<<(CCH * CS  + 255) / 256, 256, 0, stream>>>(Ww1, Ww1t, CCH, CS);

    // 2) KNN (async global->LDS staging)
    knn_kernel<<<NPTS / 256, 256, 0, stream>>>(p, idx, dist);

    // 3) fused Q/K/V projections: 1536 waves, each a 16x128 strip (32 WMMAs)
    gemm_proj_kernel<<<192, 256, 0, stream>>>(xh, Wqt_h, Wkt_h, Wvt_h,
                                              bq, bk, bv, Qp, Kp, Vp);

    // 4) fused positional encoding + score MLP + softmax + weighted sum
    attn_kernel<<<NPTS / 4, 128, 0, stream>>>(p, Qp, Kp, Vp, idx, dist,
                                              Wp1, bp1, g1, be1, Wp2, bp2,
                                              g2, be2, Ww1t, bw1, g3, be3, Ww2, bw2,
                                              out);
    (void)in_sizes; (void)n_in; (void)out_size; (void)ws_size;
}
